// Histoformer_80341658239624
// MI455X (gfx1250) — matrix operand order repaired and checked by reading, not compiled, not run on hardware
//
#include <hip/hip_runtime.h>
#include <math.h>

typedef unsigned short ushort_t;
typedef __attribute__((ext_vector_type(16))) __bf16 v16bf;
typedef __attribute__((ext_vector_type(8)))  float  v8f;

union Frag {
    v16bf v;
    ushort_t u[16];
    uint4 q[2];
};

__device__ __forceinline__ ushort_t f2bf(float f) {
    union { float f; unsigned u; } c; c.f = f;
    unsigned r = c.u + 0x7fffu + ((c.u >> 16) & 1u);   // round-to-nearest-even
    return (ushort_t)(r >> 16);
}
__device__ __forceinline__ float bf2f(ushort_t h) {
    union { unsigned u; float f; } c; c.u = ((unsigned)h) << 16;
    return c.f;
}
__device__ __forceinline__ float gelu_exact(float x) {
    return 0.5f * x * (1.0f + erff(x * 0.70710678118654752440f));
}

// ---------------------------------------------------------------------------
// Generic f32 -> bf16 cast
// ---------------------------------------------------------------------------
__global__ void cast_bf16_kernel(const float* __restrict__ in,
                                 ushort_t* __restrict__ out, int n) {
    int i = blockIdx.x * 256 + threadIdx.x;
    if (i < n) out[i] = f2bf(in[i]);
}

// ---------------------------------------------------------------------------
// Per-pixel channel LayerNorm (WithBias, no params), fp32 in -> bf16 out
// x layout [B][C][L]
// ---------------------------------------------------------------------------
__global__ void ln_kernel(const float* __restrict__ x, long xbs,
                          ushort_t* __restrict__ o, long obs, int C) {
    const int L = 65536;
    int pix = blockIdx.x * 256 + threadIdx.x;
    int b = blockIdx.y;
    const float* xp = x + b * xbs + pix;
    float s = 0.f, s2 = 0.f;
    for (int c = 0; c < C; ++c) { float v = xp[(long)c * L]; s += v; s2 += v * v; }
    float inv = 1.0f / (float)C;
    float mu = s * inv;
    float var = s2 * inv - mu * mu;
    float r = rsqrtf(var + 1e-5f);
    ushort_t* op = o + b * obs + pix;
    for (int c = 0; c < C; ++c) {
        float v = xp[(long)c * L];
        op[(long)c * L] = f2bf((v - mu) * r);
    }
}

// ---------------------------------------------------------------------------
// 3x3 depthwise conv (channel multiplier `mult`), optional exact GELU.
// Input channels 0..split-1 come from s0 (channel base s0_cb), the rest from
// s1 (supports the concat([k, feature]) branches). bf16 in/out, fp32 weights.
// ---------------------------------------------------------------------------
__launch_bounds__(256)
__global__ void dwconv3x3_kernel(const ushort_t* __restrict__ s0, long s0_bs, int s0_cb, int split,
                                 const ushort_t* __restrict__ s1, long s1_bs, int s1_cb,
                                 const float* __restrict__ w,
                                 ushort_t* __restrict__ dst, long d_bs,
                                 int mult, int do_gelu) {
    const int L = 65536;
    int pix = blockIdx.x * 256 + threadIdx.x;
    int o = blockIdx.y;
    int b = blockIdx.z;
    int cin = o / mult;
    const ushort_t* src = (cin < split)
        ? s0 + b * s0_bs + (long)(s0_cb + cin) * L
        : s1 + b * s1_bs + (long)(s1_cb + cin - split) * L;
    int y = pix >> 8, x = pix & 255;
    const float* wr = w + o * 9;
    float acc = 0.f;
#pragma unroll
    for (int dy = -1; dy <= 1; ++dy) {
        int yy = y + dy;
        if ((unsigned)yy >= 256u) continue;
#pragma unroll
        for (int dx = -1; dx <= 1; ++dx) {
            int xx = x + dx;
            if ((unsigned)xx >= 256u) continue;
            acc += wr[(dy + 1) * 3 + (dx + 1)] * bf2f(src[yy * 256 + xx]);
        }
    }
    if (do_gelu) acc = gelu_exact(acc);
    dst[b * d_bs + (long)o * L + pix] = f2bf(acc);
}

// ---------------------------------------------------------------------------
// Pointwise (1x1 conv) GEMM via WMMA bf16: Y[Cout][L] = W[Cout][Cin] * X[Cin][L]
// Block = 128 thr (4 waves). Block tile 16(M) x 256(N); each wave 16x64 (4 wmma
// per K step).
//  - W tile (16x32 = 1KB, rows contiguous) staged via ASYNC global->LDS DMA
//    (global_load_async_to_lds_b128, ASYNCcnt) issued by wave 0 only.
//  - X tile staged TRANSPOSED in LDS so B-fragments are contiguous uint4 reads.
//  - Next K-step X rows prefetched (global_prefetch_b8).
// mode: 0 = bf16 out, 1 = GELU + bf16 out, 2 = fp32 residual add, fp32 out.
// ---------------------------------------------------------------------------
__launch_bounds__(128)
__global__ void pw_gemm_kernel(const ushort_t* __restrict__ Wt, long w_bs,
                               const ushort_t* __restrict__ X, long x_bs,
                               void* __restrict__ Yv, long y_bs,
                               const float* __restrict__ R, long r_bs,
                               int Cin, int mode) {
    const int L = 65536;
    const int NBLK = 256;
    __shared__ __align__(16) ushort_t lsW[16 * 32];        // [m][k]
    __shared__ __align__(16) ushort_t lsXT[NBLK * 32];     // transposed: [col][k]

    int tid = threadIdx.x;
    int lane = tid & 31;
    int wave = tid >> 5;
    int half = lane >> 4;
    int l16 = lane & 15;
    int b = blockIdx.z;
    int m0 = blockIdx.y * 16;
    int n0 = blockIdx.x * NBLK;
    const ushort_t* Wp = Wt + b * w_bs;
    const ushort_t* Xp = X + b * x_bs;

    // Per-lane addresses for the async W-tile DMA: lane covers 32B of LDS and
    // 32B of global (row = lane>>1, 32B chunk = lane&1).
    unsigned wlds = (unsigned)(uintptr_t)(&lsW[0]) + (unsigned)lane * 32u;
    const int wrow = lane >> 1;
    const int wchk = (lane & 1) * 16;            // in elements (16 bf16 = 32B)

    const int krow = tid >> 2;                   // 0..31  (X stage)
    const int quar = tid & 3;                    // 0..3

    v8f acc[4] = {v8f{}, v8f{}, v8f{}, v8f{}};

    for (int k0 = 0; k0 < Cin; k0 += 32) {
        __syncthreads();
        // ---- async DMA the 16x32 W tile into LDS (wave 0, 32 lanes x 32B) ----
        if (wave == 0) {
            const ushort_t* g = Wp + (long)(m0 + wrow) * Cin + k0 + wchk;
            asm volatile(
                "global_load_async_to_lds_b128 %0, %1, off\n\t"
                "global_load_async_to_lds_b128 %0, %1, off offset:16"
                :: "v"(wlds), "v"(g) : "memory");
        }
        // ---- X tile: 32 rows(k) x 256 cols(n); 8 x uint4 per thread, then
        //      scatter transposed into LDS. Loads issue before any wait. ----
        {
            const uint4* s4 = (const uint4*)(Xp + (long)(k0 + krow) * L + n0 + quar * 64);
            uint4 t4[8];
#pragma unroll
            for (int j8 = 0; j8 < 8; ++j8) t4[j8] = s4[j8];
            if (k0 + 32 < Cin)   // prefetch next K-step rows
                __builtin_prefetch(Xp + (long)(k0 + 32 + krow) * L + n0 + quar * 64, 0, 1);
#pragma unroll
            for (int j8 = 0; j8 < 8; ++j8) {
                ushort_t tmp8[8];
                *(uint4*)tmp8 = t4[j8];
#pragma unroll
                for (int j = 0; j < 8; ++j)
                    lsXT[(quar * 64 + j8 * 8 + j) * 32 + krow] = tmp8[j];
            }
        }
        if (wave == 0) asm volatile("s_wait_asynccnt 0x0" ::: "memory");
        __syncthreads();

        // A fragment: 16x32 bf16. Lane holds row (lane&15); K chunks {8h..8h+7}
        // and {16+8h..16+8h+7} -> two contiguous uint4 LDS reads.
        Frag a;
        {
            const ushort_t* row = &lsW[l16 * 32];
            a.q[0] = *(const uint4*)(row + 8 * half);
            a.q[1] = *(const uint4*)(row + 16 + 8 * half);
        }
        // 4 B fragments (16 cols each): lane = col (lane&15), K = {8h..8h+7} and {16+8h..16+8h+7}
#pragma unroll
        for (int t = 0; t < 4; ++t) {
            Frag bf;
            const ushort_t* cptr = &lsXT[(wave * 64 + t * 16 + l16) * 32 + 8 * half];
            bf.q[0] = *(const uint4*)(cptr);
            bf.q[1] = *(const uint4*)(cptr + 16);
            acc[t] = __builtin_amdgcn_wmma_f32_16x16x32_bf16(
                false, a.v, false, bf.v, (short)0, acc[t], false, false);
        }
    }

    // Epilogue. D layout: col = lane&15; VGPR r -> row r + 8*(lane>=16).
#pragma unroll
    for (int t = 0; t < 4; ++t) {
        int n = n0 + wave * 64 + t * 16 + l16;
#pragma unroll
        for (int r = 0; r < 8; ++r) {
            int m = m0 + r + 8 * half;
            long idx = (long)m * L + n;
            float val = acc[t][r];
            if (mode == 0) {
                ((ushort_t*)Yv + b * y_bs)[idx] = f2bf(val);
            } else if (mode == 1) {
                ((ushort_t*)Yv + b * y_bs)[idx] = f2bf(gelu_exact(val));
            } else {
                ((float*)Yv + b * y_bs)[idx] = R[b * r_bs + idx] + val;
            }
        }
    }
}

// ---------------------------------------------------------------------------
// Gram matrix G[b,h] = Q[b,h] * K[b,h]^T  (16 x 16 over L=65536) via WMMA,
// reduction dim = pixels. Partial 16x16 tiles accumulated with fp32 atomics.
// Q rows [h*16+c][L] contiguous -> A frag = two contiguous uint4 global loads;
// K rows likewise give contiguous B frags (col = k-channel, K index = pixel).
// ---------------------------------------------------------------------------
__launch_bounds__(256)
__global__ void gram_kernel(const ushort_t* __restrict__ Q, long q_bs,
                            const ushort_t* __restrict__ K, long k_bs,
                            float* __restrict__ G) {
    const int L = 65536;
    int tid = threadIdx.x, lane = tid & 31, wave = tid >> 5;
    int half = lane >> 4, l16 = lane & 15;
    int bh = blockIdx.y;
    int b = bh / 6, h = bh % 6;
    const ushort_t* qp = Q + b * q_bs + (long)(h * 16 + l16) * L;
    const ushort_t* kp = K + b * k_bs + (long)(h * 16 + l16) * L;
    int pix0 = blockIdx.x * 1024 + wave * 128;

    v8f acc = {};
#pragma unroll
    for (int ks = 0; ks < 4; ++ks) {
        int base = pix0 + ks * 32;
        Frag a, bf;
        a.q[0] = *(const uint4*)(qp + base + 8 * half);
        a.q[1] = *(const uint4*)(qp + base + 16 + 8 * half);
        bf.q[0] = *(const uint4*)(kp + base + 8 * half);
        bf.q[1] = *(const uint4*)(kp + base + 16 + 8 * half);
        acc = __builtin_amdgcn_wmma_f32_16x16x32_bf16(
            false, a.v, false, bf.v, (short)0, acc, false, false);
    }
    float* g = G + bh * 256;
#pragma unroll
    for (int r = 0; r < 8; ++r) {
        int m = r + 8 * half;
        atomicAdd(&g[m * 16 + l16], acc[r]);
    }
}

// ---------------------------------------------------------------------------
// Row sum-of-squares (for the folded L2 norms): one block per (b, channel).
// blockIdx.y: 0 -> Q rows, 1 -> K rows.
// ---------------------------------------------------------------------------
__global__ void rowss_kernel(const ushort_t* __restrict__ Q, long q_bs,
                             const ushort_t* __restrict__ K, long k_bs,
                             float* __restrict__ qss, float* __restrict__ kss) {
    const int L = 65536;
    int row = blockIdx.x;
    int b = row / 96, c = row % 96;
    const ushort_t* p = (blockIdx.y == 0)
        ? Q + b * q_bs + (long)c * L
        : K + b * k_bs + (long)c * L;
    __shared__ float red[256];
    float s = 0.f;
    for (int l = threadIdx.x; l < L; l += 256) { float v = bf2f(p[l]); s += v * v; }
    red[threadIdx.x] = s;
    __syncthreads();
    for (int st = 128; st > 0; st >>= 1) {
        if (threadIdx.x < st) red[threadIdx.x] += red[threadIdx.x + st];
        __syncthreads();
    }
    if (threadIdx.x == 0) (blockIdx.y ? kss : qss)[row] = red[0];
}

// ---------------------------------------------------------------------------
// Softmax of scaled Gram + fold projection: Mcomb[b][i][j] =
//   sum_c proj[i][16*hj + c] * attn[b, hj, c, j%16]   (bf16 output, 96x96/batch)
// ---------------------------------------------------------------------------
__global__ void attn_finalize_kernel(const float* __restrict__ G,
                                     const float* __restrict__ qss,
                                     const float* __restrict__ kss,
                                     const float* __restrict__ temp,
                                     const float* __restrict__ proj,
                                     ushort_t* __restrict__ Mcomb) {
    int b = blockIdx.x;
    int t = threadIdx.x;
    __shared__ float attnS[6 * 16 * 16];
    if (t < 96) {
        int h = t / 16, c = t % 16;
        const float* g = G + (b * 6 + h) * 256 + c * 16;
        float rq = fmaxf(sqrtf(qss[b * 96 + h * 16 + c]), 1e-12f);
        float tm = temp[h];
        float vals[16];
        float mx = -3.0e38f;
#pragma unroll
        for (int d = 0; d < 16; ++d) {
            float rk = fmaxf(sqrtf(kss[b * 96 + h * 16 + d]), 1e-12f);
            float v = g[d] / (rq * rk) * tm;
            vals[d] = v;
            mx = fmaxf(mx, v);
        }
        float sum = 0.f;
#pragma unroll
        for (int d = 0; d < 16; ++d) { vals[d] = expf(vals[d] - mx); sum += vals[d]; }
        float inv = 1.0f / sum;
#pragma unroll
        for (int d = 0; d < 16; ++d) attnS[(h * 16 + c) * 16 + d] = vals[d] * inv;
    }
    __syncthreads();
    for (int e = t; e < 96 * 96; e += 256) {
        int i = e / 96, j = e % 96;
        int hj = j / 16, dj = j % 16;
        float s = 0.f;
#pragma unroll
        for (int c2 = 0; c2 < 16; ++c2)
            s += proj[i * 96 + hj * 16 + c2] * attnS[(hj * 16 + c2) * 16 + dj];
        Mcomb[b * 9216 + e] = f2bf(s);
    }
}

// ---------------------------------------------------------------------------
// Host launcher
// ---------------------------------------------------------------------------
extern "C" void kernel_launch(void* const* d_in, const int* in_sizes, int n_in,
                              void* d_out, int out_size, void* d_ws, size_t ws_size,
                              hipStream_t stream) {
    (void)in_sizes; (void)n_in; (void)out_size; (void)ws_size;
    const int B = 2, L = 65536;
    const long E96 = (long)B * 96 * L;     // elems
    const long E288 = (long)B * 288 * L;

    char* ws = (char*)d_ws;
    size_t off = 0;
    auto alloc = [&](size_t bytes) -> char* {
        char* p = ws + off;
        off += (bytes + 255) & ~(size_t)255;
        return p;
    };
    ushort_t* bufA  = (ushort_t*)alloc(E288 * 2);
    ushort_t* bufB  = (ushort_t*)alloc(E288 * 2);
    ushort_t* bufC  = (ushort_t*)alloc(E288 * 2);
    ushort_t* bufXN = (ushort_t*)alloc(E96 * 2);
    ushort_t* bufF1 = (ushort_t*)alloc(E96 * 2);
    ushort_t* bufF2 = (ushort_t*)alloc(E96 * 2);
    ushort_t* bufK  = (ushort_t*)alloc(E96 * 2);
    ushort_t* bufV  = (ushort_t*)alloc(E96 * 2);
    float*    bufXA = (float*)alloc(E96 * 4);
    ushort_t* w_qkv = (ushort_t*)alloc(288 * 96 * 2);
    ushort_t* w_kpw = (ushort_t*)alloc(96 * 192 * 2);
    ushort_t* w_vpw = (ushort_t*)alloc(96 * 192 * 2);
    ushort_t* w_pm  = (ushort_t*)alloc(288 * 288 * 2);
    ushort_t* w_po  = (ushort_t*)alloc(96 * 288 * 2);
    ushort_t* w_mc  = (ushort_t*)alloc((size_t)B * 96 * 96 * 2);
    float* G   = (float*)alloc((size_t)B * 6 * 256 * 4);
    float* qss = (float*)alloc((size_t)B * 96 * 4);
    float* kss = (float*)alloc((size_t)B * 96 * 4);

    const float* x       = (const float*)d_in[0];
    const float* f1      = (const float*)d_in[1];
    const float* f2      = (const float*)d_in[2];
    const float* qkv_w   = (const float*)d_in[3];
    const float* qkv_dw  = (const float*)d_in[4];
    const float* temp    = (const float*)d_in[5];
    const float* k_dw    = (const float*)d_in[6];
    const float* k_pw    = (const float*)d_in[7];
    const float* v_dw    = (const float*)d_in[8];
    const float* v_pw    = (const float*)d_in[9];
    const float* proj    = (const float*)d_in[10];
    const float* ffn_dw1 = (const float*)d_in[11];
    const float* ffn_pm  = (const float*)d_in[12];
    const float* ffn_dw2 = (const float*)d_in[13];
    const float* ffn_po  = (const float*)d_in[14];

    auto cast = [&](const float* in, ushort_t* out, int n) {
        cast_bf16_kernel<<<(n + 255) / 256, 256, 0, stream>>>(in, out, n);
    };
    cast(qkv_w, w_qkv, 288 * 96);
    cast(k_pw, w_kpw, 96 * 192);
    cast(v_pw, w_vpw, 96 * 192);
    cast(ffn_pm, w_pm, 288 * 288);
    cast(ffn_po, w_po, 96 * 288);
    cast(f1, bufF1, (int)E96);
    cast(f2, bufF2, (int)E96);

    // norm1(x) -> bf16
    ln_kernel<<<dim3(256, B), 256, 0, stream>>>(x, (long)96 * L, bufXN, (long)96 * L, 96);

    // qkv = dw3x3(pw(xn))
    pw_gemm_kernel<<<dim3(256, 18, B), 128, 0, stream>>>(
        w_qkv, 0, bufXN, (long)96 * L, bufA, (long)288 * L, nullptr, 0, 96, 0);
    dwconv3x3_kernel<<<dim3(256, 288, B), 256, 0, stream>>>(
        bufA, (long)288 * L, 0, 288, nullptr, 0, 0, qkv_dw, bufB, (long)288 * L, 1, 0);

    // k = pw(gelu(dw(concat[k, feature1])))
    dwconv3x3_kernel<<<dim3(256, 192, B), 256, 0, stream>>>(
        bufB, (long)288 * L, 96, 96, bufF1, (long)96 * L, 0, k_dw, bufC, (long)192 * L, 1, 1);
    pw_gemm_kernel<<<dim3(256, 6, B), 128, 0, stream>>>(
        w_kpw, 0, bufC, (long)192 * L, bufK, (long)96 * L, nullptr, 0, 192, 0);

    // v = pw(gelu(dw(concat[v, feature2])))
    dwconv3x3_kernel<<<dim3(256, 192, B), 256, 0, stream>>>(
        bufB, (long)288 * L, 192, 96, bufF2, (long)96 * L, 0, v_dw, bufC, (long)192 * L, 1, 1);
    pw_gemm_kernel<<<dim3(256, 6, B), 128, 0, stream>>>(
        w_vpw, 0, bufC, (long)192 * L, bufV, (long)96 * L, nullptr, 0, 192, 0);

    // Gram + row norms + softmax folded with proj into Mcomb
    hipMemsetAsync(G, 0, (size_t)B * 6 * 256 * 4, stream);
    gram_kernel<<<dim3(64, 12), 256, 0, stream>>>(bufB, (long)288 * L, bufK, (long)96 * L, G);
    rowss_kernel<<<dim3(B * 96, 2), 256, 0, stream>>>(bufB, (long)288 * L, bufK, (long)96 * L, qss, kss);
    attn_finalize_kernel<<<dim3(B), 256, 0, stream>>>(G, qss, kss, temp, proj, w_mc);

    // x_attn = x + Mcomb_b * v   (per-batch weight, fp32 out)
    pw_gemm_kernel<<<dim3(256, 6, B), 128, 0, stream>>>(
        w_mc, (long)96 * 96, bufV, (long)96 * L, bufXA, (long)96 * L, x, (long)96 * L, 96, 2);

    // FFN: ln -> dw1(mult3) -> pm(+gelu) -> dw2 -> po(+residual)
    ln_kernel<<<dim3(256, B), 256, 0, stream>>>(bufXA, (long)96 * L, bufXN, (long)96 * L, 96);
    dwconv3x3_kernel<<<dim3(256, 288, B), 256, 0, stream>>>(
        bufXN, (long)96 * L, 0, 96, nullptr, 0, 0, ffn_dw1, bufA, (long)288 * L, 3, 0);
    pw_gemm_kernel<<<dim3(256, 18, B), 128, 0, stream>>>(
        w_pm, 0, bufA, (long)288 * L, bufB, (long)288 * L, nullptr, 0, 288, 1);
    dwconv3x3_kernel<<<dim3(256, 288, B), 256, 0, stream>>>(
        bufB, (long)288 * L, 0, 288, nullptr, 0, 0, ffn_dw2, bufC, (long)288 * L, 1, 0);
    pw_gemm_kernel<<<dim3(256, 6, B), 128, 0, stream>>>(
        w_po, 0, bufC, (long)288 * L, d_out, (long)96 * L, bufXA, (long)96 * L, 288, 2);
}
